// SMRNAttnOnly_4011499454834
// MI455X (gfx1250) — matrix operand order repaired and checked by reading, not compiled, not run on hardware
//
#include <hip/hip_runtime.h>
#include <math.h>
#include <stdint.h>

// ---------------- model constants ----------------
static constexpr int B_  = 2;
static constexpr int T_  = 1024;
static constexpr int D_  = 512;
static constexpr int NF_ = 512;
static constexpr int FD_ = 1024;      // 2*NF
static constexpr int L_  = 6;
static constexpr int V_  = 32000;
static constexpr int M_  = B_ * T_;   // 2048 token rows
static constexpr float RFFS_    = 0.0625f;  // sqrt(2/512) = 1/16
static constexpr float EPS_LN_  = 1e-5f;
static constexpr float EPS_AT_  = 1e-6f;

// ---------------- WMMA / TDM types (CDNA5 wave32) ----------------
typedef __attribute__((ext_vector_type(16))) __bf16 v16bf;
typedef __attribute__((ext_vector_type(8)))  float  v8f;
typedef __attribute__((ext_vector_type(4)))  unsigned int u32x4;
typedef __attribute__((ext_vector_type(8)))  int          i32x8;
typedef __attribute__((ext_vector_type(4)))  int          i32x4;

#if __has_include(<hip/amd_detail/amd_gfx1250_TDM.h>)
#define TDM_6ARG 1
#else
#define TDM_6ARG 0
#endif

union FragBF { v16bf v; unsigned int u[8]; unsigned short h[16]; };

static __device__ inline unsigned short f2bf(float x) {
  unsigned int u = __float_as_uint(x);
  u = (u + 0x7FFFu + ((u >> 16) & 1u)) >> 16;
  return (unsigned short)u;
}
static __device__ inline float bf2f(unsigned short h) {
  return __uint_as_float(((unsigned int)h) << 16);
}
static __device__ inline float gelu_exact(float x) {
  return 0.5f * x * (1.0f + erff(x * 0.70710678118654752440f));
}

static __device__ inline v8f wmma_bf16(v16bf a, v16bf b, v8f c) {
  return __builtin_amdgcn_wmma_f32_16x16x32_bf16(
      false, a, false, b, (short)0, c, false, false);
}

// Tensor Data Mover: DMA a 2D tile of 16-bit elements (tile_d1 rows x tile_d0
// elems, row stride s0 elems in global) into LDS at lds_ptr (contiguous
// row-major). Issued once per wave; caller gates to a single wave and follows
// with s_wait_tensorcnt + a workgroup barrier.
static __device__ inline void tdm_load_2d_b16(void* lds_ptr, const void* gptr,
                                              unsigned int tensor_d0,
                                              unsigned int tensor_d1,
                                              unsigned int tile_d0,
                                              unsigned int tile_d1,
                                              unsigned int s0) {
  unsigned int lds_off = (unsigned int)(unsigned long long)(uintptr_t)lds_ptr;
  unsigned long long ga = (unsigned long long)(uintptr_t)gptr;
  u32x4 g0 = { 1u,                                 // count=1 user descriptor
               lds_off,                            // lds_addr
               (unsigned int)ga,                   // global_addr[31:0]
               (unsigned int)((ga >> 32) & 0x1FFFFFFull) | 0x80000000u }; // +type=2
  i32x8 g1 = { (int)0x00010000,                    // data_size=1 (2 bytes)
               (int)((tensor_d0 & 0xFFFFu) << 16), // tensor_dim0[15:0]
               (int)((tensor_d0 >> 16) | ((tensor_d1 & 0xFFFFu) << 16)),
               (int)((tensor_d1 >> 16) | (tile_d0 << 16)),
               (int)(tile_d1 & 0xFFFFu),           // tile_dim1 (tile_dim2=0)
               (int)s0,                            // tensor_dim0_stride[31:0]
               0, 0 };
  i32x4 z4 = { 0, 0, 0, 0 };
#if TDM_6ARG
  i32x8 z8 = { 0, 0, 0, 0, 0, 0, 0, 0 };
  __builtin_amdgcn_tensor_load_to_lds(g0, g1, z4, z4, z8, 0);
#else
  __builtin_amdgcn_tensor_load_to_lds(g0, g1, z4, z4, 0);
#endif
}

// A-operand frag (16x32 bf16) from LDS row-major [row][k], K-pairs contiguous.
static __device__ inline v16bf frag_A_lds(const unsigned short* lds, int ld,
                                          int row0, int k0, int lane) {
  FragBF f; int half = (lane >> 4) & 1; int r = lane & 15;
  const unsigned short* base = lds + (row0 + r) * ld + k0;
#pragma unroll
  for (int v = 0; v < 8; ++v) {
    int k = half * 16 + v * 2;   // half*16+v*2
    f.u[v] = *(const unsigned int*)(base + k);
  }
  return f.v;
}

// B-operand frag (32x16 bf16) from LDS stored TRANSPOSED as [n][k].
static __device__ inline v16bf frag_B_ldsT(const unsigned short* lds, int ld,
                                           int n0, int k0, int lane) {
  FragBF f; int half = (lane >> 4) & 1; int c = lane & 15;
  const unsigned short* base = lds + (n0 + c) * ld + k0 + half * 16;
#pragma unroll
  for (int v = 0; v < 8; ++v) f.u[v] = *(const unsigned int*)(base + 2 * v);
  return f.v;
}

// A-operand frag from f32 LDS [row][k] (converted to bf16 on the fly).
static __device__ inline v16bf frag_A_f32(const float* lds, int ld,
                                          int row0, int k0, int lane) {
  FragBF f; int half = (lane >> 4) & 1; int r = lane & 15;
  const float* base = lds + (row0 + r) * ld + k0;
#pragma unroll
  for (int v = 0; v < 8; ++v) {
    int k = half * 16 + v * 2;
    f.h[2 * v]     = f2bf(base[k]);
    f.h[2 * v + 1] = f2bf(base[k + 1]);
  }
  return f.v;
}

// B-operand frag from f32 LDS stored [k][n] (the running state S).
static __device__ inline v16bf frag_B_f32(const float* lds, int ld,
                                          int k0, int n0, int lane) {
  FragBF f; int half = (lane >> 4) & 1; int c = lane & 15;
  const float* base = lds + (k0 + half * 16) * ld + n0 + c;
#pragma unroll
  for (int v = 0; v < 8; ++v) {
    f.h[2 * v]     = f2bf(base[(2 * v) * ld]);
    f.h[2 * v + 1] = f2bf(base[(2 * v + 1) * ld]);
  }
  return f.v;
}

// ---------------- generic bf16 GEMM, 128x64x64 block tile, 8 waves ----------------
#define GBM 128
#define GBN 64
#define GBK 64
#define FUSE_BIAS 1
#define FUSE_GELU 2
#define FUSE_RES  4

__global__ __launch_bounds__(256) void gemm_bf16_kernel(
    const unsigned short* __restrict__ A,   // (M,K) row-major bf16
    const unsigned short* __restrict__ B,   // (K,N) or (N,K) if transB
    const float* __restrict__ bias,
    const float* __restrict__ resid,
    float* __restrict__ Cf, unsigned short* __restrict__ Cb,
    int M, int N, int K, int transB, int fuse)
{
  __shared__ unsigned short Alds[GBM * GBK];   // [m][k], TDM-filled
  __shared__ unsigned short Blds[GBN * GBK];   // [n][k]
  int tid = threadIdx.x, lane = tid & 31, wave = tid >> 5;
  int wm = wave & 3, wn = wave >> 2;           // wave grid 4(M) x 2(N)
  int m0 = blockIdx.y * GBM, n0 = blockIdx.x * GBN;

  v8f acc[2][2];
#pragma unroll
  for (int i = 0; i < 2; ++i)
#pragma unroll
    for (int j = 0; j < 2; ++j)
#pragma unroll
      for (int e = 0; e < 8; ++e) acc[i][j][e] = 0.f;

  for (int k0 = 0; k0 < K; k0 += GBK) {
    // A tile via Tensor Data Mover: 128 rows x 64 elems, row stride K.
    if (tid < 32) {
      tdm_load_2d_b16(Alds, A + (size_t)m0 * K + k0,
                      (unsigned)K, (unsigned)M, GBK, GBM, (unsigned)K);
      __builtin_amdgcn_s_wait_tensorcnt((short)0);
    }
    if (!transB) {
      // B global [K][N]: transpose into LDS [n][k]
#pragma unroll
      for (int i = 0; i < 16; ++i) {
        int e = tid + i * 256;
        int kk = e & 63, nn = e >> 6;
        Blds[nn * GBK + kk] = B[(size_t)(k0 + kk) * N + n0 + nn];
      }
    } else {
      // B global [N][K]: contiguous dword copy
#pragma unroll
      for (int i = 0; i < 8; ++i) {
        int e = tid + i * 256;
        int nn = e >> 5, kp = e & 31;
        *(unsigned int*)(Blds + nn * GBK + kp * 2) =
            *(const unsigned int*)(B + (size_t)(n0 + nn) * K + k0 + kp * 2);
      }
    }
    if (k0 + GBK < K)  // hint next B rows into cache (global_prefetch_b8)
      __builtin_prefetch(B + (size_t)(k0 + GBK + (tid & 63)) * (transB ? K : N), 0, 1);
    __syncthreads();

#pragma unroll
    for (int kk = 0; kk < GBK; kk += 32) {
      v16bf a0 = frag_A_lds(Alds, GBK, wm * 32,      kk, lane);
      v16bf a1 = frag_A_lds(Alds, GBK, wm * 32 + 16, kk, lane);
      v16bf b0 = frag_B_ldsT(Blds, GBK, wn * 32,      kk, lane);
      v16bf b1 = frag_B_ldsT(Blds, GBK, wn * 32 + 16, kk, lane);
      acc[0][0] = wmma_bf16(a0, b0, acc[0][0]);
      acc[0][1] = wmma_bf16(a0, b1, acc[0][1]);
      acc[1][0] = wmma_bf16(a1, b0, acc[1][0]);
      acc[1][1] = wmma_bf16(a1, b1, acc[1][1]);
    }
    __syncthreads();
  }

  int half = (lane >> 4) & 1, c = lane & 15;
#pragma unroll
  for (int tm = 0; tm < 2; ++tm)
#pragma unroll
    for (int tn = 0; tn < 2; ++tn)
#pragma unroll
      for (int i = 0; i < 8; ++i) {
        int gm = m0 + wm * 32 + tm * 16 + i + half * 8;
        int gn = n0 + wn * 32 + tn * 16 + c;
        if (gm < M && gn < N) {
          float v = acc[tm][tn][i];
          if (fuse & FUSE_BIAS) v += bias[gn];
          if (fuse & FUSE_GELU) v = gelu_exact(v);
          size_t idx = (size_t)gm * N + gn;
          if (fuse & FUSE_RES)  v += resid[idx];
          if (Cf) Cf[idx] = v;
          if (Cb) Cb[idx] = f2bf(v);
        }
      }
}

// ---------------- embedding + positional ----------------
__global__ __launch_bounds__(256) void embed_kernel(
    const int* __restrict__ tok, const float* __restrict__ emb,
    const float* __restrict__ pos, float* __restrict__ xf,
    unsigned short* __restrict__ xb)
{
  int row = blockIdx.x;            // b*T + t
  int t = row & (T_ - 1);
  int v = tok[row];
  for (int d = threadIdx.x; d < D_; d += 256) {
    float val = emb[(size_t)v * D_ + d] + pos[(size_t)t * D_ + d];
    xf[(size_t)row * D_ + d] = val;
    xb[(size_t)row * D_ + d] = f2bf(val);
  }
}

// ---------------- f32 -> bf16 ----------------
__global__ void cvt_bf16_kernel(const float* __restrict__ in,
                                unsigned short* __restrict__ out, int n)
{
  int i = blockIdx.x * blockDim.x + threadIdx.x;
  if (i < n) out[i] = f2bf(in[i]);
}

// ---------------- RFF feature map: phi = s*[cos(P), sin(P)] ----------------
__global__ __launch_bounds__(256) void phi_kernel(
    const float* __restrict__ P, unsigned short* __restrict__ phib)
{
  int row = blockIdx.x;
  const float* pr = P + (size_t)row * NF_;
  unsigned short* o = phib + (size_t)row * FD_;
  for (int j = threadIdx.x; j < NF_; j += 256) {
    float p = pr[j];
    o[j]       = f2bf(RFFS_ * cosf(p));
    o[NF_ + j] = f2bf(RFFS_ * sinf(p));
  }
}

// ---------------- layernorm ----------------
__global__ __launch_bounds__(256) void ln_kernel(
    const float* __restrict__ x, const float* __restrict__ g,
    const float* __restrict__ bta, float* __restrict__ of,
    unsigned short* __restrict__ ob)
{
  __shared__ float rs[256], rq[256];
  int row = blockIdx.x;
  const float* xr = x + (size_t)row * D_;
  float s = 0.f, q = 0.f;
  for (int d = threadIdx.x; d < D_; d += 256) { float v = xr[d]; s += v; q += v * v; }
  rs[threadIdx.x] = s; rq[threadIdx.x] = q;
  __syncthreads();
  for (int off = 128; off > 0; off >>= 1) {
    if (threadIdx.x < off) {
      rs[threadIdx.x] += rs[threadIdx.x + off];
      rq[threadIdx.x] += rq[threadIdx.x + off];
    }
    __syncthreads();
  }
  float mu  = rs[0] * (1.0f / D_);
  float var = rq[0] * (1.0f / D_) - mu * mu;
  float inv = rsqrtf(var + EPS_LN_);
  for (int d = threadIdx.x; d < D_; d += 256) {
    float v = (xr[d] - mu) * inv * g[d] + bta[d];
    of[(size_t)row * D_ + d] = v;
    ob[(size_t)row * D_ + d] = f2bf(v);
  }
}

// ---------------- chunked causal linear attention ----------------
// Grid: (D/DVS, B). Each block owns S[:, d0:d0+DVS] (f32, resident in LDS
// across all chunks) + its own copy of Z. Chunks of 64 timesteps; all heavy
// math is bf16 WMMA with f32 accumulation; Q/K strips staged by the TDM.
static constexpr int CHK = 64;   // chunk length
static constexpr int DVS = 32;   // dv slice per block
static constexpr int FS  = 128;  // feature-dim strip

static constexpr size_t SM_S    = (size_t)FD_ * DVS * 4;   // 131072
static constexpr size_t SM_Z    = (size_t)FD_ * 4;         // 4096
static constexpr size_t SM_QS   = (size_t)CHK * FS * 2;    // 16384
static constexpr size_t SM_KS   = SM_QS;
static constexpr size_t SM_KT   = (size_t)FS * CHK * 2;    // 16384
static constexpr size_t SM_VT   = (size_t)DVS * CHK * 2;   // 4096
static constexpr size_t SM_ALOC = (size_t)CHK * CHK * 4;   // 16384
static constexpr size_t SM_DEN  = (size_t)CHK * 4;
static constexpr size_t SM_RED  = 256 * 4;
static constexpr size_t ATTN_SMEM =
    SM_S + SM_Z + SM_QS + SM_KS + SM_KT + SM_VT + SM_ALOC + SM_DEN + SM_RED;

__global__ __launch_bounds__(256) void linattn_kernel(
    const unsigned short* __restrict__ Qp,  // (B,T,FD) bf16
    const unsigned short* __restrict__ Kp,  // (B,T,FD) bf16
    const unsigned short* __restrict__ Vb,  // (B,T,D)  bf16
    unsigned short* __restrict__ Yb)        // (B,T,D)  bf16
{
  extern __shared__ char smem[];
  float*          S    = (float*)smem;                            // [FD][DVS]
  float*          Z    = (float*)(smem + SM_S);                   // [FD]
  unsigned short* Qs   = (unsigned short*)(smem + SM_S + SM_Z);   // [CHK][FS]
  unsigned short* Ks   = Qs + CHK * FS;                           // [CHK][FS]
  unsigned short* Kt   = Ks + CHK * FS;                           // [FS][CHK]
  unsigned short* Vt   = Kt + FS * CHK;                           // [DVS][CHK]
  float*          Aloc = (float*)(Vt + DVS * CHK);                // [CHK][CHK]
  float*          den  = Aloc + CHK * CHK;                        // [CHK]
  float*          red  = den + CHK;                               // [256]

  int tid = threadIdx.x, lane = tid & 31, wave = tid >> 5;
  int b = blockIdx.y, d0 = blockIdx.x * DVS;

  for (int i = tid; i < FD_ * DVS; i += 256) S[i] = 0.f;
  for (int i = tid; i < FD_;       i += 256) Z[i] = 0.f;

  int tt = wave & 3, td = wave >> 2;          // output tile (64x32 -> 4x2 tiles)
  int at1 = wave & 3, at2 = (wave >> 2) * 2;  // Aloc tiles (4x4 grid, 2/wave)

  for (int c0 = 0; c0 < T_; c0 += CHK) {
    __syncthreads();   // prior chunk fully consumed LDS buffers

    // stage V^T slice: Vt[d][t]
    for (int e = tid; e < DVS * CHK; e += 256) {
      int d = e >> 6, t = e & 63;
      Vt[d * CHK + t] = Vb[((size_t)b * T_ + c0 + t) * D_ + d0 + d];
    }

    v8f yacc, aacc0, aacc1;
#pragma unroll
    for (int e = 0; e < 8; ++e) { yacc[e] = 0.f; aacc0[e] = 0.f; aacc1[e] = 0.f; }
    float denp = 0.f;

    for (int fs = 0; fs < FD_; fs += FS) {
      // stage Qp/Kp strips [CHK][FS] via Tensor Data Mover (wave 0 issues)
      if (tid < 32) {
        const size_t gofs = ((size_t)b * T_ + c0) * FD_ + fs;
        tdm_load_2d_b16(Qs, Qp + gofs, FD_, T_, FS, CHK, FD_);
        tdm_load_2d_b16(Ks, Kp + gofs, FD_, T_, FS, CHK, FD_);
        __builtin_amdgcn_s_wait_tensorcnt((short)0);
      }
      __syncthreads();
      // build transposed Kt[f][t] from Ks
      for (int e = tid; e < FS * CHK; e += 256) {
        int f = e >> 6, t = e & 63;
        Kt[f * CHK + t] = Ks[t * FS + f];
      }
      __syncthreads();

      // (a) inter-chunk numerator: yacc += Qs_strip @ bf16(S_strip)
#pragma unroll
      for (int kk = 0; kk < FS; kk += 32) {
        v16bf a  = frag_A_lds(Qs, FS, tt * 16, kk, lane);
        v16bf bb = frag_B_f32(S + (size_t)fs * DVS, DVS, kk, td * 16, lane);
        yacc = wmma_bf16(a, bb, yacc);
      }
      // (b) intra-chunk scores: aacc += Qs_strip @ Ks_strip^T
#pragma unroll
      for (int kk = 0; kk < FS; kk += 32) {
        v16bf a  = frag_A_lds(Qs, FS, at1 * 16, kk, lane);
        v16bf b0 = frag_B_ldsT(Ks, FS, at2 * 16,      kk, lane);
        v16bf b1 = frag_B_ldsT(Ks, FS, at2 * 16 + 16, kk, lane);
        aacc0 = wmma_bf16(a, b0, aacc0);
        aacc1 = wmma_bf16(a, b1, aacc1);
      }
      // inter-chunk denominator partial (reads Z before this strip is updated)
      {
        int row = tid >> 2, part = tid & 3;
        const unsigned short* qr = Qs + row * FS + part * 32;
        const float* zr = Z + fs + part * 32;
        float p = 0.f;
#pragma unroll
        for (int j = 0; j < 32; ++j) p += bf2f(qr[j]) * zr[j];
        denp += p;
      }
      __syncthreads();

      // (c) state update for this strip: S[fs:fs+FS, :] += Kt @ V_slice
      {
        int frow = fs + wave * 16;
        v16bf a0 = frag_A_lds(Kt, CHK, wave * 16, 0,  lane);
        v16bf a1 = frag_A_lds(Kt, CHK, wave * 16, 32, lane);
        int half = (lane >> 4) & 1, cc = lane & 15;
#pragma unroll
        for (int dt = 0; dt < 2; ++dt) {
          float* Stile = S + (size_t)frow * DVS + dt * 16;
          v8f sc;
#pragma unroll
          for (int i = 0; i < 8; ++i) sc[i] = Stile[(i + half * 8) * DVS + cc];
          v16bf b0 = frag_B_ldsT(Vt, CHK, dt * 16, 0,  lane);
          v16bf b1 = frag_B_ldsT(Vt, CHK, dt * 16, 32, lane);
          sc = wmma_bf16(a0, b0, sc);
          sc = wmma_bf16(a1, b1, sc);
#pragma unroll
          for (int i = 0; i < 8; ++i) Stile[(i + half * 8) * DVS + cc] = sc[i];
        }
      }
      // Z strip update: Z[f] += sum_t Kp[t][f]
      if (tid < FS) {
        const unsigned short* kr = Kt + tid * CHK;
        float s = 0.f;
#pragma unroll
        for (int t = 0; t < CHK; ++t) s += bf2f(kr[t]);
        Z[fs + tid] += s;
      }
      __syncthreads();
    } // feature strips

    // causal mask + write scores to LDS
    {
      int half = (lane >> 4) & 1, cc = lane & 15;
#pragma unroll
      for (int i = 0; i < 8; ++i) {
        int r  = at1 * 16 + i + half * 8;
        int c1 = at2 * 16 + cc;
        int c2 = c1 + 16;
        Aloc[r * CHK + c1] = (c1 <= r) ? aacc0[i] : 0.f;
        Aloc[r * CHK + c2] = (c2 <= r) ? aacc1[i] : 0.f;
      }
    }
    red[tid] = denp;
    __syncthreads();
    // den[t] = qp.Z_prev + rowsum(masked scores)
    if (tid < CHK) {
      float s = red[tid * 4] + red[tid * 4 + 1] + red[tid * 4 + 2] + red[tid * 4 + 3];
      const float* ar = Aloc + tid * CHK;
#pragma unroll
      for (int j = 0; j < CHK; ++j) s += ar[j];
      den[tid] = s;
    }
    __syncthreads();

    // intra-chunk: yacc += mask(A) @ V_slice
#pragma unroll
    for (int kk = 0; kk < CHK; kk += 32) {
      v16bf a  = frag_A_f32(Aloc, CHK, tt * 16, kk, lane);
      v16bf bb = frag_B_ldsT(Vt, CHK, td * 16, kk, lane);
      yacc = wmma_bf16(a, bb, yacc);
    }
    // normalize + store
    {
      int half = (lane >> 4) & 1, cc = lane & 15;
#pragma unroll
      for (int i = 0; i < 8; ++i) {
        int r    = tt * 16 + i + half * 8;
        int dcol = td * 16 + cc;
        float o = yacc[i] / (den[r] + EPS_AT_);
        Yb[((size_t)b * T_ + c0 + r) * D_ + d0 + dcol] = f2bf(o);
      }
    }
  } // chunks
}

// ---------------- host driver ----------------
extern "C" void kernel_launch(void* const* d_in, const int* in_sizes, int n_in,
                              void* d_out, int out_size, void* d_ws, size_t ws_size,
                              hipStream_t stream) {
  (void)in_sizes; (void)n_in; (void)out_size; (void)ws_size;
  const int*   tokens = (const int*)d_in[0];
  const float* embed  = (const float*)d_in[1];
  const float* pos    = (const float*)d_in[2];
  const float* Wq     = (const float*)d_in[3];
  const float* Wk     = (const float*)d_in[4];
  const float* Wv     = (const float*)d_in[5];
  const float* Wo     = (const float*)d_in[6];
  const float* omega  = (const float*)d_in[7];
  const float* ln_g   = (const float*)d_in[8];
  const float* ln_b   = (const float*)d_in[9];
  const float* W1     = (const float*)d_in[10];
  const float* b1     = (const float*)d_in[11];
  const float* W2     = (const float*)d_in[12];
  const float* b2     = (const float*)d_in[13];
  const float* lnf_g  = (const float*)d_in[14];
  const float* lnf_b  = (const float*)d_in[15];
  float* out = (float*)d_out;

  char* wsp = (char*)d_ws;
  size_t off = 0;
  auto alloc = [&](size_t bytes) -> char* {
    char* p = wsp + off; off += (bytes + 255) & ~(size_t)255; return p;
  };

  float*          xf   = (float*)alloc((size_t)M_ * D_ * 4);
  unsigned short* xb   = (unsigned short*)alloc((size_t)M_ * D_ * 2);
  unsigned short* eb   = (unsigned short*)alloc((size_t)V_ * D_ * 2);
  unsigned short* wqb  = (unsigned short*)alloc((size_t)D_ * D_ * 2);
  unsigned short* wkb  = (unsigned short*)alloc((size_t)D_ * D_ * 2);
  unsigned short* wvb  = (unsigned short*)alloc((size_t)D_ * D_ * 2);
  unsigned short* wob  = (unsigned short*)alloc((size_t)D_ * D_ * 2);
  unsigned short* womb = (unsigned short*)alloc((size_t)D_ * NF_ * 2);
  unsigned short* w1b  = (unsigned short*)alloc((size_t)D_ * 4 * D_ * 2);
  unsigned short* w2b  = (unsigned short*)alloc((size_t)4 * D_ * D_ * 2);
  unsigned short* Qb   = (unsigned short*)alloc((size_t)M_ * D_ * 2);
  unsigned short* Kb   = (unsigned short*)alloc((size_t)M_ * D_ * 2);
  unsigned short* Vbuf = (unsigned short*)alloc((size_t)M_ * D_ * 2);
  float*          Pf   = (float*)alloc((size_t)M_ * NF_ * 4);
  unsigned short* Qpb  = (unsigned short*)alloc((size_t)M_ * FD_ * 2);
  unsigned short* Kpb  = (unsigned short*)alloc((size_t)M_ * FD_ * 2);
  unsigned short* Yb   = (unsigned short*)alloc((size_t)M_ * D_ * 2);
  float*          xres = (float*)alloc((size_t)M_ * D_ * 4);
  unsigned short* hb   = (unsigned short*)alloc((size_t)M_ * 4 * D_ * 2);
  unsigned short* lnb  = (unsigned short*)alloc((size_t)M_ * D_ * 2);

  auto cvt = [&](const float* src, unsigned short* dst, int n) {
    cvt_bf16_kernel<<<(n + 255) / 256, 256, 0, stream>>>(src, dst, n);
  };
  auto gemm = [&](const unsigned short* Ab, const unsigned short* Bb,
                  const float* bias, const float* resid, float* Cf,
                  unsigned short* Cb, int Mm, int Nn, int Kk, int transB, int fuse) {
    dim3 g((Nn + GBN - 1) / GBN, (Mm + GBM - 1) / GBM);
    gemm_bf16_kernel<<<g, 256, 0, stream>>>(Ab, Bb, bias, resid, Cf, Cb,
                                            Mm, Nn, Kk, transB, fuse);
  };

  // x = embed[tokens] + pos
  embed_kernel<<<M_, 256, 0, stream>>>(tokens, embed, pos, xf, xb);
  cvt(embed, eb, V_ * D_);

  for (int l = 0; l < L_; ++l) {
    cvt(Wq + (size_t)l * D_ * D_, wqb, D_ * D_);
    cvt(Wk + (size_t)l * D_ * D_, wkb, D_ * D_);
    cvt(Wv + (size_t)l * D_ * D_, wvb, D_ * D_);
    cvt(Wo + (size_t)l * D_ * D_, wob, D_ * D_);
    cvt(omega + (size_t)l * D_ * NF_, womb, D_ * NF_);
    cvt(W1 + (size_t)l * D_ * 4 * D_, w1b, D_ * 4 * D_);
    cvt(W2 + (size_t)l * 4 * D_ * D_, w2b, 4 * D_ * D_);

    // projections
    gemm(xb, wqb, nullptr, nullptr, nullptr, Qb,   M_, D_, D_, 0, 0);
    gemm(xb, wkb, nullptr, nullptr, nullptr, Kb,   M_, D_, D_, 0, 0);
    gemm(xb, wvb, nullptr, nullptr, nullptr, Vbuf, M_, D_, D_, 0, 0);

    // RFF features
    gemm(Qb, womb, nullptr, nullptr, Pf, nullptr, M_, NF_, D_, 0, 0);
    phi_kernel<<<M_, 256, 0, stream>>>(Pf, Qpb);
    gemm(Kb, womb, nullptr, nullptr, Pf, nullptr, M_, NF_, D_, 0, 0);
    phi_kernel<<<M_, 256, 0, stream>>>(Pf, Kpb);

    // chunked causal linear attention
    linattn_kernel<<<dim3(D_ / DVS, B_), 256, ATTN_SMEM, stream>>>(Qpb, Kpb, Vbuf, Yb);

    // output proj + residual, then LN
    gemm(Yb, wob, nullptr, xf, xres, nullptr, M_, D_, D_, 0, FUSE_RES);
    ln_kernel<<<M_, 256, 0, stream>>>(xres, ln_g + (size_t)l * D_,
                                      ln_b + (size_t)l * D_, xf, xb);

    // FFN
    gemm(xb, w1b, b1 + (size_t)l * 4 * D_, nullptr, nullptr, hb,
         M_, 4 * D_, D_, 0, FUSE_BIAS | FUSE_GELU);
    gemm(hb, w2b, b2 + (size_t)l * D_, xf, xf, xb,
         M_, D_, 4 * D_, 0, FUSE_BIAS | FUSE_RES);
  }

  // final LN + tied lm_head (x @ embed^T)
  ln_kernel<<<M_, 256, 0, stream>>>(xf, lnf_g, lnf_b, xres, lnb);
  gemm(lnb, eb, nullptr, nullptr, out, nullptr, M_, V_, D_, 1, 0);
}
